// FP4Net_62251255988412
// MI455X (gfx1250) — matrix operand order repaired and checked by reading, not compiled, not run on hardware
//
#include <hip/hip_runtime.h>

// ---------------------------------------------------------------------------
// FP4 (bnb E2M1) 4-layer MLP for MI455X (gfx1250, wave32, WMMA).
//   h = relu(x @ dq(w1).T + b1); ... ; out = sigmoid(h3 @ dq(w4).T + b4)
// v3: peeled 3-stage async pipeline -> branch-free steady-state loop:
//     4x global_load_async_to_lds_b128 | 12x ds_load_b128 | 8x wmma |
//     s_wait_asynccnt 4 | barrier, plus a 2-iteration drain tail.
// ---------------------------------------------------------------------------

typedef __bf16 bf16;
typedef __attribute__((ext_vector_type(8)))  __bf16 v8bf;
typedef __attribute__((ext_vector_type(16))) __bf16 v16bf;
typedef __attribute__((ext_vector_type(8)))  float  v8f;

// ----------------------------- fp4 dequant ---------------------------------
__device__ __forceinline__ float fp4_nearest(float n) {
  float a  = __builtin_fabsf(n);
  float bc = 0.0f;      // best code
  float bd = a;         // distance to code 0.0
#define FP4_TRY(c) { float d = __builtin_fabsf(a - (c)); if (d < bd) { bd = d; bc = (c); } }
  FP4_TRY(0.0052083333f)
  FP4_TRY(0.1666667f)
  FP4_TRY(0.25f)
  FP4_TRY(0.3333333f)
  FP4_TRY(0.5f)
  FP4_TRY(0.6666667f)
  FP4_TRY(1.0f)
#undef FP4_TRY
  return __builtin_copysignf(bc, n);
}

__global__ __launch_bounds__(256) void fp4_dequant_kernel(
    const float* __restrict__ w, bf16* __restrict__ out, int nblocks) {
  int blk = blockIdx.x * blockDim.x + threadIdx.x;
  if (blk >= nblocks) return;
  const float4* p = (const float4*)(w + (size_t)blk * 64);
  float4 v[16];
  float amax = 0.0f;
#pragma unroll
  for (int i = 0; i < 16; ++i) {
    v[i] = p[i];
    amax = fmaxf(amax,
           fmaxf(fmaxf(__builtin_fabsf(v[i].x), __builtin_fabsf(v[i].y)),
                 fmaxf(__builtin_fabsf(v[i].z), __builtin_fabsf(v[i].w))));
  }
  float inv = (amax == 0.0f) ? 1.0f : (1.0f / amax);
  bf16* o = out + (size_t)blk * 64;
#pragma unroll
  for (int i = 0; i < 8; ++i) {
    float4 a = v[2 * i], b = v[2 * i + 1];
    v8bf q;
    q[0] = (bf16)(fp4_nearest(a.x * inv) * amax);
    q[1] = (bf16)(fp4_nearest(a.y * inv) * amax);
    q[2] = (bf16)(fp4_nearest(a.z * inv) * amax);
    q[3] = (bf16)(fp4_nearest(a.w * inv) * amax);
    q[4] = (bf16)(fp4_nearest(b.x * inv) * amax);
    q[5] = (bf16)(fp4_nearest(b.y * inv) * amax);
    q[6] = (bf16)(fp4_nearest(b.z * inv) * amax);
    q[7] = (bf16)(fp4_nearest(b.w * inv) * amax);
    *(v8bf*)(o + i * 8) = q;
  }
}

// ------------------------- f32 -> bf16 convert ------------------------------
__global__ __launch_bounds__(256) void f32_to_bf16_kernel(
    const float* __restrict__ x, bf16* __restrict__ out, int n) {
  int i = (blockIdx.x * blockDim.x + threadIdx.x) * 8;
  if (i >= n) return;
  float4 a = *(const float4*)(x + i);
  float4 b = *(const float4*)(x + i + 4);
  v8bf q;
  q[0] = (bf16)a.x; q[1] = (bf16)a.y; q[2] = (bf16)a.z; q[3] = (bf16)a.w;
  q[4] = (bf16)b.x; q[5] = (bf16)b.y; q[6] = (bf16)b.z; q[7] = (bf16)b.w;
  *(v8bf*)(out + i) = q;
}

// ------------------------------- WMMA GEMM ----------------------------------
// C[M,N] = act(A[M,K] @ W[N,K]^T + bias[N]); both operands K-contiguous.
// 128x128 tile per 256-thread block (8 waves, each wave: 32x64 = 2x4 WMMA
// tiles of 16x16), BK=32. 3-stage LDS pipeline fed by async global->LDS
// copies (ASYNCcnt), so tile kt+2 streams in while tiles kt/kt+1 compute.
// Requires K % 32 == 0 and K/32 >= 3 (true for all four layers here).
#define BM 128
#define BN 128
#define BK 32
#define NSTAGE 3

template <bool FINAL>
__global__ __launch_bounds__(256) void gemm_wmma_kernel(
    const bf16* __restrict__ A, const bf16* __restrict__ W,
    const float* __restrict__ bias, void* __restrict__ outp,
    int M, int N, int K) {
  __shared__ __attribute__((aligned(32))) bf16 As[NSTAGE][BM * BK];
  __shared__ __attribute__((aligned(32))) bf16 Ws[NSTAGE][BN * BK];

  const int tid  = threadIdx.x;
  const int lane = tid & 31;        // wave32
  const int wid  = tid >> 5;        // 0..7
  const int wm   = wid & 3;         // 32-row slab within tile
  const int wn   = wid >> 2;        // 64-col slab within tile
  const int l    = lane & 15;
  const int hb   = lane >> 4;       // half-wave select

  const int m0 = blockIdx.y * BM;
  const int n0 = blockIdx.x * BN;

  // global->LDS async copy mapping: 256 threads x 2 rows x 16B cover each
  // 128x32 bf16 tile (4 x b128 async copies per thread per stage).
  const int lrow = tid >> 2;          // 0..63
  const int lcol = (tid & 3) * 8;     // 0,8,16,24 (bf16 elems)

  // LDS byte addresses (low 32 bits of the generic pointer = LDS offset).
  const unsigned ldsA = (unsigned)(uintptr_t)&As[0][0] + (unsigned)(lrow * BK + lcol) * 2u;
  const unsigned ldsW = (unsigned)(uintptr_t)&Ws[0][0] + (unsigned)(lrow * BK + lcol) * 2u;
  const unsigned halfTile = (unsigned)(64 * BK) * 2u;       // +64 rows in LDS
  const unsigned bufB     = (unsigned)(BM * BK) * 2u;       // stage stride

  // Global byte offsets (GVS mode: saddr base + 32-bit vaddr offset).
  const unsigned oA = (unsigned)(((size_t)(m0 + lrow) * K + lcol) * 2u);
  const unsigned oW = (unsigned)(((size_t)(n0 + lrow) * K + lcol) * 2u);
  const unsigned gRow64 = (unsigned)((size_t)64 * K * 2u);

  auto fetch_async = [&](int kt, int buf) {
    const unsigned kb = (unsigned)kt * (BK * 2u);
    const unsigned bo = (unsigned)buf * bufB;
    asm volatile("global_load_async_to_lds_b128 %0, %1, %2"
                 :: "v"(ldsA + bo), "v"(oA + kb), "s"(A) : "memory");
    asm volatile("global_load_async_to_lds_b128 %0, %1, %2"
                 :: "v"(ldsA + bo + halfTile), "v"(oA + gRow64 + kb), "s"(A) : "memory");
    asm volatile("global_load_async_to_lds_b128 %0, %1, %2"
                 :: "v"(ldsW + bo), "v"(oW + kb), "s"(W) : "memory");
    asm volatile("global_load_async_to_lds_b128 %0, %1, %2"
                 :: "v"(ldsW + bo + halfTile), "v"(oW + gRow64 + kb), "s"(W) : "memory");
  };

  v8f acc[2][4];
#pragma unroll
  for (int i = 0; i < 2; ++i)
#pragma unroll
    for (int j = 0; j < 4; ++j)
#pragma unroll
      for (int r = 0; r < 8; ++r) acc[i][j][r] = 0.0f;

  // A fragment (16x32, M x K), contiguous K order matching B:
  // lanes 0-15: K=0..15; lanes 16-31: K=16..31
  auto load_a = [&](int buf, int row16) -> v16bf {
    return *(const v16bf*)&As[buf][(row16 + l) * BK + hb * 16];
  };
  // B fragment (32x16, K x N): lane n holds 16 consecutive K values of column
  // n (= row n of W, K-contiguous) -> one contiguous 32B LDS read per lane.
  auto load_b = [&](int buf, int row16) -> v16bf {
    return *(const v16bf*)&Ws[buf][(row16 + l) * BK + hb * 16];
  };

  auto compute = [&](int buf) {
    v16bf af[2], wf[4];
#pragma unroll
    for (int tm = 0; tm < 2; ++tm) af[tm] = load_a(buf, wm * 32 + tm * 16);
#pragma unroll
    for (int tn = 0; tn < 4; ++tn) wf[tn] = load_b(buf, wn * 64 + tn * 16);
#pragma unroll
    for (int tm = 0; tm < 2; ++tm)
#pragma unroll
      for (int tn = 0; tn < 4; ++tn)
        acc[tm][tn] = __builtin_amdgcn_wmma_f32_16x16x32_bf16(
            false, af[tm], false, wf[tn], (short)0, acc[tm][tn], false, false);
  };

  const int ksteps = K / BK;   // >= 3 for all layers
  fetch_async(0, 0);
  fetch_async(1, 1);
  // Stage 0 landed when at most the newest group of 4 remains (in-order).
  asm volatile("s_wait_asynccnt 0x4" ::: "memory");
  __syncthreads();

  // Steady state: branch-free pipeline body.
  int kt = 0;
  for (; kt + 2 < ksteps; ++kt) {
    fetch_async(kt + 2, (kt + 2) % NSTAGE);
    compute(kt % NSTAGE);
    // All but the group just issued (tile kt+2) are complete -> tile kt+1 is
    // ready for every wave after the barrier.
    asm volatile("s_wait_asynccnt 0x4" ::: "memory");
    __syncthreads();
  }
  // Drain: 2 remaining tiles, nothing left in flight after wait 0.
  for (; kt < ksteps; ++kt) {
    compute(kt % NSTAGE);
    asm volatile("s_wait_asynccnt 0x0" ::: "memory");
    __syncthreads();
  }

  // Epilogue: C/D layout -> VGPR r: lanes 0-15 hold M=r, lanes 16-31 hold M=8+r
  float* outf = (float*)outp;
  bf16*  outb = (bf16*)outp;
#pragma unroll
  for (int tm = 0; tm < 2; ++tm) {
    const int mb = m0 + wm * 32 + tm * 16 + hb * 8;
#pragma unroll
    for (int tn = 0; tn < 4; ++tn) {
      const int gn = n0 + wn * 64 + tn * 16 + l;
      const float bv = bias[gn];
      size_t idx = (size_t)mb * N + gn;
#pragma unroll
      for (int r = 0; r < 8; ++r) {
        float v = acc[tm][tn][r] + bv;
        if (FINAL) outf[idx] = 1.0f / (1.0f + __expf(-v));   // sigmoid
        else       outb[idx] = (bf16)fmaxf(v, 0.0f);          // relu -> bf16
        idx += (size_t)N;
      }
    }
  }
}

// ------------------------------- launcher -----------------------------------
extern "C" void kernel_launch(void* const* d_in, const int* in_sizes, int n_in,
                              void* d_out, int out_size, void* d_ws, size_t ws_size,
                              hipStream_t stream) {
  const float* x  = (const float*)d_in[0];
  const float* w1 = (const float*)d_in[1];
  const float* b1 = (const float*)d_in[2];
  const float* w2 = (const float*)d_in[3];
  const float* b2 = (const float*)d_in[4];
  const float* w3 = (const float*)d_in[5];
  const float* b3 = (const float*)d_in[6];
  const float* w4 = (const float*)d_in[7];
  const float* b4 = (const float*)d_in[8];

  const int H   = in_sizes[2];            // 4096
  const int IN  = in_sizes[1] / H;        // 1024
  const int Bm  = in_sizes[0] / IN;       // 8192
  const int OUT = in_sizes[8];            // 1024

  // Workspace layout (bytes): xb | hA | hB | wdq   (~176 MB total)
  char* ws = (char*)d_ws;
  bf16* xb  = (bf16*)ws;
  bf16* hA  = (bf16*)(ws + (size_t)Bm * IN * 2);
  bf16* hB  = (bf16*)(ws + (size_t)Bm * IN * 2 + (size_t)Bm * H * 2);
  bf16* wdq = (bf16*)(ws + (size_t)Bm * IN * 2 + (size_t)Bm * H * 4);

  // x -> bf16
  {
    int n = Bm * IN;
    f32_to_bf16_kernel<<<(n / 8 + 255) / 256, 256, 0, stream>>>(x, xb, n);
  }

  dim3 blk(256);

  // Layer 1: [Bm,IN] @ [H,IN]^T -> hA
  {
    int nb = (H * IN) / 64;
    fp4_dequant_kernel<<<(nb + 255) / 256, blk, 0, stream>>>(w1, wdq, nb);
    gemm_wmma_kernel<false><<<dim3(H / BN, Bm / BM), blk, 0, stream>>>(
        xb, wdq, b1, hA, Bm, H, IN);
  }
  // Layer 2: hA @ [H,H]^T -> hB
  {
    int nb = (H * H) / 64;
    fp4_dequant_kernel<<<(nb + 255) / 256, blk, 0, stream>>>(w2, wdq, nb);
    gemm_wmma_kernel<false><<<dim3(H / BN, Bm / BM), blk, 0, stream>>>(
        hA, wdq, b2, hB, Bm, H, H);
  }
  // Layer 3: hB @ [H,H]^T -> hA
  {
    int nb = (H * H) / 64;
    fp4_dequant_kernel<<<(nb + 255) / 256, blk, 0, stream>>>(w3, wdq, nb);
    gemm_wmma_kernel<false><<<dim3(H / BN, Bm / BM), blk, 0, stream>>>(
        hB, wdq, b3, hA, Bm, H, H);
  }
  // Layer 4: hA @ [OUT,H]^T -> sigmoid -> d_out (f32)
  {
    int nb = (OUT * H) / 64;
    fp4_dequant_kernel<<<(nb + 255) / 256, blk, 0, stream>>>(w4, wdq, nb);
    gemm_wmma_kernel<true><<<dim3(OUT / BN, Bm / BM), blk, 0, stream>>>(
        hA, wdq, b4, d_out, Bm, OUT, H);
  }
}
